// bilateralFilter_58780922413106
// MI455X (gfx1250) — matrix operand order repaired and checked by reading, not compiled, not run on hardware
//
#include <hip/hip_runtime.h>

// ---------------- problem constants (fixed by the reference) ----------------
#define WINR   3            // window radius
#define CH     16           // channels / embedding features
#define IMH    256
#define IMW    256
#define PXW    16           // pixels per wave (one WMMA M dim)
#define WAVES  4            // waves per block
#define PXB    (PXW*WAVES)  // 64 pixels per block (one row segment)
#define RCOLS  (PXB + 2*WINR)   // 70 staged columns
#define RROWS  (2*WINR + 1)     // 7 staged rows
#define NB     ((2*WINR+1)*(PXW + 2*WINR))  // 7*22 = 154 neighbors per wave tile
#define NBP    160          // padded to 5 K-tiles of 32

typedef __attribute__((ext_vector_type(16))) _Float16 v16h;
typedef __attribute__((ext_vector_type(8)))  _Float16 v8h;
typedef __attribute__((ext_vector_type(8)))  float    v8f;

// wave32 xor-butterfly within each 16-lane half via ds_swizzle (group-of-32,
// and_mask=0x1f, xor_mask in bits [14:10]); xor masks <=8 never cross bit 4.
#define SWZ_XOR(v, imm) \
  __builtin_bit_cast(float, __builtin_amdgcn_ds_swizzle(__builtin_bit_cast(int, (v)), (imm)))

__global__ __launch_bounds__(128)
void bilateralFilter_58780922413106_kernel(const float* __restrict__ embed,
                                           const float* __restrict__ xin,
                                           float* __restrict__ out)
{
    __shared__ __align__(16) _Float16 Elds[RROWS * RCOLS * CH];
    __shared__ __align__(16) _Float16 Xlds[RROWS * RCOLS * CH];
    __shared__ float                  Nrm [RROWS * RCOLS];
    __shared__ __align__(16) _Float16 Wlds[WAVES * PXW * NBP];

    const int tid  = threadIdx.x;
    const int lane = tid & 31;
    const int wv   = tid >> 5;       // wave id 0..3
    const int hl   = lane >> 4;      // half id (0: lanes 0-15, 1: lanes 16-31)
    const int ln   = lane & 15;

    int bid = blockIdx.x;
    const int wseg = bid & 3;  bid >>= 2;      // IMW / PXB = 4 segments per row
    const int row  = bid & (IMH - 1); bid >>= 8;
    const int n    = bid;                      // batch
    const int wb   = wseg * PXB;               // base output column

    const size_t plane = (size_t)IMH * IMW;    // 65536
    const float* eb = embed + (size_t)n * CH * plane;
    const float* xb = xin   + (size_t)n * CH * plane;
    float*       ob = out   + (size_t)n * CH * plane;

    // ---- stage embed & x region (f16, zero-padded) into LDS ----
    for (int i = tid; i < RROWS * RCOLS * CH; i += 128) {
        int col = i % RCOLS;
        int t   = i / RCOLS;
        int ch  = t % CH;
        int dy  = t / CH;
        int gr  = row + dy - WINR;
        int gc  = wb  + col - WINR;
        float ev = 0.f, xv = 0.f;
        if ((unsigned)gr < IMH && (unsigned)gc < IMW) {
            size_t a = (size_t)ch * plane + (size_t)gr * IMW + gc;
            ev = eb[a];
            xv = xb[a];
        }
        int o = (dy * RCOLS + col) * CH + ch;
        Elds[o] = (_Float16)ev;
        Xlds[o] = (_Float16)xv;
    }
    __syncthreads();

    // ---- per-neighbor squared norms (fp32, from the f16-rounded values) ----
    for (int i = tid; i < RROWS * RCOLS; i += 128) {
        const _Float16* p = &Elds[i * CH];
        float s = 0.f;
        #pragma unroll
        for (int c = 0; c < CH; ++c) { float v = (float)p[c]; s += v * v; }
        Nrm[i] = s;
    }
    __syncthreads();

    // ---- A matrix: center embeddings, 16 px x K(=16 of 32) f16 ----
    // A layout: lane m=ln; element e<8 -> K = e + 8*hl (channel); e>=8 -> K>=16 (zero)
    v16h amat;
    {
        int ccol = WINR + wv * PXW + ln;    // region column of this lane's center
        v8h e8 = *(const v8h*)&Elds[(WINR * RCOLS + ccol) * CH + 8 * hl];
        #pragma unroll
        for (int e = 0; e < 8; ++e) { amat[e] = e8[e]; amat[e + 8] = (_Float16)0.f; }
    }

    // center norms for the 8 M-rows this lane's half owns (M = r + 8*hl)
    float cn[8];
    #pragma unroll
    for (int r = 0; r < 8; ++r)
        cn[r] = Nrm[WINR * RCOLS + WINR + wv * PXW + r + 8 * hl];

    float wsum[8];
    #pragma unroll
    for (int r = 0; r < 8; ++r) wsum[r] = 0.f;

    const float VCOEF = 50.0f;                 // 0.5 / sigma_v^2, sigma_v = 0.1
    const float SCOEF = 1.0204081632653061f;   // 0.5 / sigma_s^2, sigma_s = 0.7

    // ---- GEMM1: dot(center, neighbor) over 10 N-tiles of 16 neighbors ----
    #pragma unroll
    for (int nt = 0; nt < 10; ++nt) {
        int j  = nt * 16 + ln;                 // this lane's neighbor column N
        int jc = j > NB - 1 ? NB - 1 : j;
        int dy = jc / 22;
        int dx = jc - dy * 22;                 // 0..21 within wave region
        int rcol = wv * PXW + dx;

        // B layout: N = ln; element e<8 -> K = e + 8*hl (channel); e>=8 -> K>=16 (zero)
        v16h bmat;
        {
            const v8h* p0 = (const v8h*)&Elds[(dy * RCOLS + rcol) * CH];
            v8h b0 = p0[hl];
            #pragma unroll
            for (int e = 0; e < 8; ++e) { bmat[e] = b0[e]; bmat[e + 8] = (_Float16)0.f; }
        }

        v8f cz = {};
        v8f d = __builtin_amdgcn_wmma_f32_16x16x32_f16(
                    false, amat, false, bmat, (short)0, cz, false, false);

        float nn  = Nrm[dy * RCOLS + rcol];
        float dyv = (float)(dy - WINR);
        #pragma unroll
        for (int r = 0; r < 8; ++r) {
            int m   = r + 8 * hl;                        // D row: M = r + 8*(lane>=16)
            int dxr = dx - WINR - m;                     // neighbor x - center x
            float dist2 = fmaf(-2.f, d[r], cn[r] + nn);
            dist2 = dist2 < 0.f ? 0.f : dist2;
            float w = __expf(-VCOEF * dist2 - SCOEF * ((float)(dxr * dxr) + dyv * dyv));
            bool ok = (dxr >= -WINR) & (dxr <= WINR) & (j < NB);
            w = ok ? w : 0.f;
            wsum[r] += w;
            Wlds[(wv * PXW + m) * NBP + j] = (_Float16)w;
        }
    }

    // ---- weight row sums: butterfly over the 16 lanes of each half ----
    #pragma unroll
    for (int r = 0; r < 8; ++r) {
        float s = wsum[r];
        s += SWZ_XOR(s, 0x041F);   // xor 1
        s += SWZ_XOR(s, 0x081F);   // xor 2
        s += SWZ_XOR(s, 0x101F);   // xor 4
        s += SWZ_XOR(s, 0x201F);   // xor 8
        wsum[r] = s + 1e-20f;      // + reg_constant
    }

    // ---- GEMM2: result[16 px x 16 ch] = W[16 x 160] * X[160 x 16] ----
    v8f acc = {};
    #pragma unroll
    for (int kt = 0; kt < 5; ++kt) {
        // A = W: lane M = ln; e<8 -> K = kt*32 + e + 8*hl ; e>=8 -> K = kt*32 + 16 + (e-8) + 8*hl
        v16h wa;
        {
            int base = (wv * PXW + ln) * NBP + kt * 32 + 8 * hl;
            v8h w0 = *(const v8h*)&Wlds[base];
            v8h w1 = *(const v8h*)&Wlds[base + 16];
            #pragma unroll
            for (int e = 0; e < 8; ++e) { wa[e] = w0[e]; wa[e + 8] = w1[e]; }
        }
        // B = X: lane N = ln (channel); e<8 -> K = kt*32 + e + 8*hl ; e>=8 -> K = kt*32 + 16 + (e-8) + 8*hl (neighbor j)
        v16h xbm;
        #pragma unroll
        for (int e = 0; e < 16; ++e) {
            int j  = kt * 32 + (e & 7) + 8 * hl + 16 * (e >> 3);
            int jc = j > NB - 1 ? NB - 1 : j;   // W is 0 there; value irrelevant
            int dy = jc / 22;
            int dx = jc - dy * 22;
            xbm[e] = Xlds[(dy * RCOLS + wv * PXW + dx) * CH + ln];
        }
        acc = __builtin_amdgcn_wmma_f32_16x16x32_f16(
                  false, wa, false, xbm, (short)0, acc, false, false);
    }

    // ---- normalize and store: D row M = r + 8*hl, channel = ln ----
    #pragma unroll
    for (int r = 0; r < 8; ++r) {
        int m  = r + 8 * hl;
        int px = wb + wv * PXW + m;
        ob[(size_t)ln * plane + (size_t)row * IMW + px] = acc[r] / wsum[r];
    }
}

extern "C" void kernel_launch(void* const* d_in, const int* in_sizes, int n_in,
                              void* d_out, int out_size, void* d_ws, size_t ws_size,
                              hipStream_t stream) {
    (void)n_in; (void)out_size; (void)d_ws; (void)ws_size;
    const float* embed = (const float*)d_in[0];
    const float* xin   = (const float*)d_in[1];
    float*       out   = (float*)d_out;

    const int n_img  = in_sizes[0] / (CH * IMH * IMW);   // 2
    const int blocks = n_img * IMH * (IMW / PXB);        // 2*256*4 = 2048
    hipLaunchKernelGGL(bilateralFilter_58780922413106_kernel,
                       dim3(blocks), dim3(128), 0, stream, embed, xin, out);
}
